// FrozenLLMEncoder_59227599012004
// MI455X (gfx1250) — compile-verified, matrix operand order checked by reading, not run on hardware
//
#include <hip/hip_runtime.h>
#include <hip/hip_bf16.h>
#include <math.h>

// ---------------------------------------------------------------------------
// FrozenLLMEncoder forward for MI455X (gfx1250, wave32, WMMA).
// GEMMs + attention on v_wmma_f32_16x16x32_bf16 (f32 accumulate); bf16 A tiles
// staged into LDS via GLOBAL_LOAD_ASYNC_TO_LDS_B128 (ASYNCcnt path).
// ---------------------------------------------------------------------------

typedef __attribute__((ext_vector_type(16))) __bf16 v16bf;
typedef __attribute__((ext_vector_type(8)))  float  v8f;

#define DEV __device__ __forceinline__

DEV v8f wmma_bf16(v16bf a, v16bf b, v8f c) {
  // 8 args: (neg_a, A, neg_b, B, c_mod, C, reuse_a, reuse_b)
  return __builtin_amdgcn_wmma_f32_16x16x32_bf16(false, a, false, b,
                                                 (short)0, c, false, false);
}

// gfx1250 async copy: global -> LDS, 16B per lane, tracked by ASYNCcnt.
// ldsOff is the workgroup-relative LDS byte offset (low 32 bits of the generic
// address of a __shared__ object); gaddr is the per-lane global address.
DEV void async_load_b128(unsigned ldsOff, const void* gaddr) {
  asm volatile("global_load_async_to_lds_b128 %0, %1, off"
               :: "v"(ldsOff), "v"(gaddr) : "memory");
}
DEV void wait_async0() {
  asm volatile("s_wait_asynccnt 0" ::: "memory");
}

// Per-lane fragment element e (=2j+t) maps to K = base(j) + 8*half + t where
// base(j) = j<4 ? 2j : 16+2(j-4)  (ISA 7.12.2, 16-bit A 16x32 layout; B uses
// the same pattern with lane%16 = column).

// Fragment from an LDS row with contiguous K (row start 16B-aligned).
DEV v16bf ldsfrag(const __bf16* rowPtr, int half) {
  union { uint4 u; __bf16 h[8]; } lo, hi;
  lo.u = *(const uint4*)(rowPtr + 8 * half);
  hi.u = *(const uint4*)(rowPtr + 16 + 8 * half);
  v16bf f;
#pragma unroll
  for (int e = 0; e < 8; ++e) { f[e] = lo.h[e]; f[8 + e] = hi.h[e]; }
  return f;
}

// Fragment from a global fp32 row (K contiguous), converting to bf16 on load.
DEV v16bf gfrag(const float* row, int half) {
  v16bf f;
#pragma unroll
  for (int j = 0; j < 8; ++j) {
    const int d0 = ((j < 4) ? 2 * j : 16 + 2 * (j - 4)) + 8 * half;
    const float2 v = *(const float2*)(row + d0);
    f[2 * j]     = (__bf16)v.x;
    f[2 * j + 1] = (__bf16)v.y;
  }
  return f;
}

// ---------------------------------------------------------------------------
// GEMM: C[M,N] = A[M,K](bf16) * W[K,N](fp32->bf16 on the fly) + bias
// MODE 0: OutF  = C                  (qkv)
// MODE 1: OutF += C                  (residual add into h: attn proj, fc2)
// MODE 2: OutB  = bf16(gelu(C))      (fc1, exact erf gelu)
// 256 threads = 8 waves; 128x128 macro-tile, BK=32. Wave w owns a 32x64
// region: rows (w&3)*32 + {0,16}, cols (w>>2)*64 + {0..3}*16 -> 8 accumulators,
// 8 WMMAs per K-step from 2 A-frags x 4 B-frags.
// A tile (128x32 bf16, 8KB) arrives via async-to-LDS; W tile (32x128 f32)
// through VGPRs (needs f32->bf16 convert) and is stored transposed.
// ---------------------------------------------------------------------------
template <int MODE>
__global__ __launch_bounds__(256)
void gemm_wmma_k(const __bf16* __restrict__ A, const float* __restrict__ W,
                 const float* __restrict__ bias, float* __restrict__ OutF,
                 __bf16* __restrict__ OutB, int N, int K)
{
  __shared__ alignas(16) __bf16 sA[128][40];    // [m][k], padded rows (80B)
  __shared__ alignas(16) __bf16 sBT[128][40];   // [n][k] transposed weights
  const int tid  = threadIdx.x;
  const int lane = tid & 31, wave = tid >> 5;
  const int n    = lane & 15, half = lane >> 4;
  const int wRow = (wave & 3) * 32;
  const int wCol = (wave >> 2) * 64;
  const int mBase = blockIdx.y * 128;
  const int nBase = blockIdx.x * 128;

  // A staging: thread -> 16 bf16: row tid/2, cols (tid&1)*16 + {0..15}
  const int ar = tid >> 1, ac = (tid & 1) * 16;
  const unsigned aLds = (unsigned)(size_t)(&sA[ar][ac]);
  // W staging: thread -> 16 f32: row tid/8, cols (tid&7)*16 + {0..15}
  const int br = tid >> 3, bc = (tid & 7) * 16;

  v8f acc[8] = {};

  for (int k0 = 0; k0 < K; k0 += 32) {
    const float* wp = W + (size_t)(k0 + br) * N + nBase + bc;
    const float4 w0 = *(const float4*)(wp + 0);
    const float4 w1 = *(const float4*)(wp + 4);
    const float4 w2 = *(const float4*)(wp + 8);
    const float4 w3 = *(const float4*)(wp + 12);
    if (k0 + 32 < K)  // global_prefetch_b8 of the next weight tile
      __builtin_prefetch(W + (size_t)(k0 + 32 + br) * N + nBase + bc, 0, 1);
    __syncthreads();  // previous K-step's fragment reads done; LDS reusable
    const __bf16* ap = A + (size_t)(mBase + ar) * K + k0 + ac;
    async_load_b128(aLds,      ap);
    async_load_b128(aLds + 16, ap + 8);
    const float wv[16] = {w0.x, w0.y, w0.z, w0.w, w1.x, w1.y, w1.z, w1.w,
                          w2.x, w2.y, w2.z, w2.w, w3.x, w3.y, w3.z, w3.w};
#pragma unroll
    for (int q = 0; q < 16; ++q) sBT[bc + q][br] = (__bf16)wv[q];
    wait_async0();    // this wave's async LDS writes complete
    __syncthreads();  // all waves' tiles visible
    v16bf af[2], bfr[4];
#pragma unroll
    for (int s = 0; s < 2; ++s) af[s] = ldsfrag(&sA[wRow + s * 16 + n][0], half);
#pragma unroll
    for (int c = 0; c < 4; ++c) bfr[c] = ldsfrag(&sBT[wCol + c * 16 + n][0], half);
#pragma unroll
    for (int s = 0; s < 2; ++s)
#pragma unroll
      for (int c = 0; c < 4; ++c)
        acc[s * 4 + c] = wmma_bf16(af[s], bfr[c], acc[s * 4 + c]);
  }

#pragma unroll
  for (int s = 0; s < 2; ++s)
#pragma unroll
    for (int c = 0; c < 4; ++c) {
      const v8f a = acc[s * 4 + c];
      const int col = nBase + wCol + c * 16 + n;
      const float bv = bias[col];
#pragma unroll
      for (int i = 0; i < 8; ++i) {
        const int row = mBase + wRow + s * 16 + i + 8 * half;  // D: M=i+8*half
        const float val = a[i] + bv;
        if constexpr (MODE == 0) {
          OutF[(size_t)row * N + col] = val;
        } else if constexpr (MODE == 1) {
          OutF[(size_t)row * N + col] += val;
        } else {
          const float g = 0.5f * val * (1.0f + erff(val * 0.70710678118654752f));
          OutB[(size_t)row * N + col] = (__bf16)g;
        }
      }
    }
}

// ---------------------------------------------------------------------------
// LayerNorm over D=1024; one block per token. OUTF=0 -> bf16 (GEMM A input),
// OUTF=1 -> f32 (end-of-block LN overwrites h in place).
// ---------------------------------------------------------------------------
template <int OUTF>
__global__ __launch_bounds__(256)
void layernorm_k(const float* __restrict__ x, const float* __restrict__ gs,
                 const float* __restrict__ gb, __bf16* __restrict__ outB,
                 float* __restrict__ outF)
{
  const int row = blockIdx.x;
  const int tid = threadIdx.x;
  const int lane = tid & 31, wave = tid >> 5;
  const float* xr = x + (size_t)row * 1024;
  float v[4], sum = 0.f, sq = 0.f;
#pragma unroll
  for (int i = 0; i < 4; ++i) {
    v[i] = xr[tid + 256 * i];
    sum += v[i]; sq += v[i] * v[i];
  }
#pragma unroll
  for (int off = 16; off > 0; off >>= 1) {
    sum += __shfl_down(sum, off);
    sq  += __shfl_down(sq,  off);
  }
  __shared__ float rS[8], rQ[8], stats[2];
  if (lane == 0) { rS[wave] = sum; rQ[wave] = sq; }
  __syncthreads();
  if (tid == 0) {
    float ts = 0.f, tq = 0.f;
#pragma unroll
    for (int w = 0; w < 8; ++w) { ts += rS[w]; tq += rQ[w]; }
    const float mean = ts * (1.0f / 1024.0f);
    const float var  = tq * (1.0f / 1024.0f) - mean * mean;
    stats[0] = mean;
    stats[1] = rsqrtf(var + 1e-5f);
  }
  __syncthreads();
  const float mean = stats[0], rstd = stats[1];
#pragma unroll
  for (int i = 0; i < 4; ++i) {
    const int d = tid + 256 * i;
    const float y = (v[i] - mean) * rstd * gs[d] + gb[d];
    if constexpr (OUTF) outF[(size_t)row * 1024 + d] = y;
    else                outB[(size_t)row * 1024 + d] = (__bf16)y;
  }
}

// ---------------------------------------------------------------------------
// Flash attention: one wave per (b, head, 16-query tile). Online softmax over
// chunks of 32 keys; scores via WMMA (K=64 head dim), P.V via WMMA (K=32 keys).
// qkv layout per token: [H=16][q 64 | k 64 | v 64]. ctx written as bf16 [B*T,D].
// ---------------------------------------------------------------------------
__global__ __launch_bounds__(32)
void attn_wmma_k(const float* __restrict__ qkv, const float* __restrict__ mask,
                 __bf16* __restrict__ ctx)
{
  const int T = 1024;
  const int qt = blockIdx.x, hh = blockIdx.y, b = blockIdx.z;
  const int lane = threadIdx.x;
  const int n = lane & 15, half = lane >> 4;
  __shared__ alignas(16) float  sS[16][36];
  __shared__ alignas(16) __bf16 sP[16][40];
  __shared__ float sAlpha[16];
  __shared__ float sL[16];

  const float* qrow = qkv + ((size_t)(b * T + qt * 16 + n) * 3072) + hh * 192;
  const v16bf qf0 = gfrag(qrow, half);        // d 0..31
  const v16bf qf1 = gfrag(qrow + 32, half);   // d 32..63

  float m_run = -INFINITY, l_run = 0.f;       // valid in lanes 0..15 (row=lane)
  v8f o[4] = {};                              // 16x64 output accumulators

  for (int kt = 0; kt < 32; ++kt) {           // 32 chunks of 32 keys
    const int kbase = kt * 32;
#pragma unroll
    for (int sub = 0; sub < 2; ++sub) {
      const int krow = kbase + sub * 16 + n;  // B-frag column = key index
      const float* krp = qkv + ((size_t)(b * T + krow) * 3072) + hh * 192 + 64;
      const v16bf kf0 = gfrag(krp, half);
      const v16bf kf1 = gfrag(krp + 32, half);
      v8f s = {};
      s = wmma_bf16(qf0, kf0, s);
      s = wmma_bf16(qf1, kf1, s);
      const float lm = logf(fmaxf(mask[b * T + krow], 1e-9f));
#pragma unroll
      for (int i = 0; i < 8; ++i)
        sS[i + 8 * half][sub * 16 + n] = s[i] * 0.125f + lm;
    }
    __syncthreads();
    if (lane < 16) {                          // per-row online softmax
      const int r = lane;
      float sv[32], cm = -INFINITY;
#pragma unroll
      for (int c = 0; c < 32; ++c) { sv[c] = sS[r][c]; cm = fmaxf(cm, sv[c]); }
      const float mn    = fmaxf(m_run, cm);
      const float alpha = __expf(m_run - mn);
      float ps = 0.f;
#pragma unroll
      for (int c = 0; c < 32; ++c) {
        const float p = __expf(sv[c] - mn);
        ps += p;
        sP[r][c] = (__bf16)p;
      }
      l_run = l_run * alpha + ps;
      m_run = mn;
      sAlpha[r] = alpha;
    }
    __syncthreads();
    float al[8];
#pragma unroll
    for (int i = 0; i < 8; ++i) al[i] = sAlpha[i + 8 * half];
    const v16bf pf = ldsfrag(&sP[n][0], half);  // A-frag: P 16x32
#pragma unroll
    for (int c = 0; c < 4; ++c) {
#pragma unroll
      for (int i = 0; i < 8; ++i) o[c][i] *= al[i];
      v16bf vf;                                 // B-frag: V 32keys x 16 d-cols
#pragma unroll
      for (int j = 0; j < 8; ++j) {
        const int kb = ((j < 4) ? 2 * j : 16 + 2 * (j - 4)) + 8 * half;
        const float* vp = qkv + ((size_t)(b * T + kbase + kb) * 3072) +
                          hh * 192 + 128 + c * 16 + n;
        vf[2 * j]     = (__bf16)vp[0];
        vf[2 * j + 1] = (__bf16)vp[3072];       // next key row
      }
      o[c] = wmma_bf16(pf, vf, o[c]);
    }
    __syncthreads();
  }
  if (lane < 16) sL[lane] = l_run;
  __syncthreads();
  float linv[8];
#pragma unroll
  for (int i = 0; i < 8; ++i) linv[i] = 1.0f / sL[i + 8 * half];
#pragma unroll
  for (int c = 0; c < 4; ++c)
#pragma unroll
    for (int i = 0; i < 8; ++i) {
      const int row = qt * 16 + i + 8 * half;
      const int col = hh * 64 + c * 16 + n;
      ctx[(size_t)(b * T + row) * 1024 + col] = (__bf16)(o[c][i] * linv[i]);
    }
}

// ---------------------------------------------------------------------------
// Embedding gather, and masked mean-pool + L2 normalize.
// ---------------------------------------------------------------------------
__global__ __launch_bounds__(256)
void embed_k(const int* __restrict__ ids, const float* __restrict__ emb,
             float* __restrict__ h)
{
  const int tok = blockIdx.x;
  const int id  = ids[tok];
  const float4* src = (const float4*)(emb + (size_t)id * 1024);
  float4* dst = (float4*)(h + (size_t)tok * 1024);
  dst[threadIdx.x] = src[threadIdx.x];
}

__global__ __launch_bounds__(256)
void pool_norm_k(const float* __restrict__ h, const float* __restrict__ pm,
                 float* __restrict__ out)
{
  const int b = blockIdx.x, tid = threadIdx.x;
  const int lane = tid & 31, wave = tid >> 5;
  float acc[4] = {0.f, 0.f, 0.f, 0.f};
  float msum = 0.f;
  for (int t = 0; t < 1024; ++t) {
    const float w = pm[b * 1024 + t];
    msum += w;
    const float* hr = h + (size_t)(b * 1024 + t) * 1024;
#pragma unroll
    for (int i = 0; i < 4; ++i) acc[i] += hr[tid + 256 * i] * w;
  }
  const float inv = 1.0f / fmaxf(msum, 1e-6f);
  float s[4], sq = 0.f;
#pragma unroll
  for (int i = 0; i < 4; ++i) { s[i] = acc[i] * inv; sq += s[i] * s[i]; }
#pragma unroll
  for (int off = 16; off > 0; off >>= 1) sq += __shfl_down(sq, off);
  __shared__ float red[8];
  if (lane == 0) red[wave] = sq;
  __syncthreads();
  if (tid == 0) {
    float t = 0.f;
#pragma unroll
    for (int w = 0; w < 8; ++w) t += red[w];
    red[0] = t;
  }
  __syncthreads();
  const float nrm = sqrtf(red[0]);
  const float r = 1.0f / fmaxf(nrm, 1e-12f);
#pragma unroll
  for (int i = 0; i < 4; ++i) out[b * 1024 + tid + 256 * i] = s[i] * r;
}

// ---------------------------------------------------------------------------
// Driver. Workspace layout (112 MB):
//   h    f32 [4096,1024]   @ 0        (16 MB)
//   qkv  f32 [4096,3072]   @ 16 MB    (48 MB)
//   xb   bf16[4096,1024]   @ 64 MB    (8 MB)   (LN output -> GEMM A)
//   ctx  bf16[4096,1024]   @ 72 MB    (8 MB)
//   y1b  bf16[4096,4096]   @ 80 MB    (32 MB)  (gelu output -> fc2 A)
// ---------------------------------------------------------------------------
extern "C" void kernel_launch(void* const* d_in, const int* in_sizes, int n_in,
                              void* d_out, int out_size, void* d_ws, size_t ws_size,
                              hipStream_t stream)
{
  const int*   ids   = (const int*)  d_in[0];
  const float* amask = (const float*)d_in[1];
  const float* pmask = (const float*)d_in[2];
  const float* emb   = (const float*)d_in[3];
  const float* ln1s  = (const float*)d_in[4];
  const float* ln1b  = (const float*)d_in[5];
  const float* qkvw  = (const float*)d_in[6];
  const float* qkvb  = (const float*)d_in[7];
  const float* attnw = (const float*)d_in[8];
  const float* attnb = (const float*)d_in[9];
  const float* ln2s  = (const float*)d_in[10];
  const float* ln2b  = (const float*)d_in[11];
  const float* fc1w  = (const float*)d_in[12];
  const float* fc1b  = (const float*)d_in[13];
  const float* fc2w  = (const float*)d_in[14];
  const float* fc2b  = (const float*)d_in[15];
  float* out = (float*)d_out;

  char* ws = (char*)d_ws;
  float*  h   = (float*) (ws);
  float*  qkv = (float*) (ws + (size_t)16 * 1024 * 1024);
  __bf16* xb  = (__bf16*)(ws + (size_t)64 * 1024 * 1024);
  __bf16* ctx = (__bf16*)(ws + (size_t)72 * 1024 * 1024);
  __bf16* y1b = (__bf16*)(ws + (size_t)80 * 1024 * 1024);

  embed_k<<<4096, 256, 0, stream>>>(ids, emb, h);

  for (int l = 0; l < 6; ++l) {
    layernorm_k<0><<<4096, 256, 0, stream>>>(h, ln1s + l * 1024, ln1b + l * 1024, xb, nullptr);
    gemm_wmma_k<0><<<dim3(24, 32), 256, 0, stream>>>(
        xb, qkvw + (size_t)l * 1024 * 3072, qkvb + l * 3072, qkv, nullptr, 3072, 1024);
    attn_wmma_k<<<dim3(64, 16, 4), 32, 0, stream>>>(qkv, amask, ctx);
    gemm_wmma_k<1><<<dim3(8, 32), 256, 0, stream>>>(
        ctx, attnw + (size_t)l * 1024 * 1024, attnb + l * 1024, h, nullptr, 1024, 1024);
    layernorm_k<0><<<4096, 256, 0, stream>>>(h, ln2s + l * 1024, ln2b + l * 1024, xb, nullptr);
    gemm_wmma_k<2><<<dim3(32, 32), 256, 0, stream>>>(
        xb, fc1w + (size_t)l * 1024 * 4096, fc1b + l * 4096, nullptr, y1b, 4096, 1024);
    gemm_wmma_k<1><<<dim3(8, 32), 256, 0, stream>>>(
        y1b, fc2w + (size_t)l * 4096 * 1024, fc2b + l * 1024, h, nullptr, 1024, 4096);
    layernorm_k<1><<<4096, 256, 0, stream>>>(h, ln2s + l * 1024, ln2b + l * 1024, nullptr, h);
  }

  pool_norm_k<<<4, 256, 0, stream>>>(h, pmask, out);

  (void)in_sizes; (void)n_in; (void)out_size; (void)ws_size;
}
